// Net_3762391351854
// MI455X (gfx1250) — matrix operand order repaired and checked by reading, not compiled, not run on hardware
//
#include <hip/hip_runtime.h>
#include <stdint.h>
#include <stddef.h>

// ---------------------------------------------------------------------------
// Problem constants (from reference)
// ---------------------------------------------------------------------------
static constexpr int Nn    = 2048;   // nodes
static constexpr int Ee    = 6144;   // edges
static constexpr int DVIN  = 128;
static constexpr int DVOUT = 64;
static constexpr int DEIN  = 16;
static constexpr int DEOUT = 16;

#define DEV __device__ __forceinline__

typedef __bf16 bf16_t;
typedef __attribute__((ext_vector_type(16))) __bf16 v16bf;
typedef __attribute__((ext_vector_type(8)))  __bf16 v8bf;
typedef __attribute__((ext_vector_type(8)))  float  v8f;
typedef __attribute__((ext_vector_type(2)))  float  v2f;

#if defined(__has_builtin)
#  if __has_builtin(__builtin_amdgcn_wmma_f32_16x16x4_f32)
#    define HAVE_WMMA_F32 1
#  endif
#endif
#ifndef HAVE_WMMA_F32
#  define HAVE_WMMA_F32 0
#endif

// float -> bf16 (round to nearest even), stored as raw u16
DEV uint16_t f2bf_raw(float f) {
  union { float f; uint32_t u; } x; x.f = f;
  uint32_t r = x.u + 0x7FFFu + ((x.u >> 16) & 1u);
  return (uint16_t)(r >> 16);
}

// ---------------------------------------------------------------------------
// WMMA bf16 fragment loaders (wave32; same contiguous k order for A and B)
// A 16x32 (row-major source, ld=K): lane&15 = row; per-lane K-chunks
//   elements 0..7  -> k = (lane>>4)*16 + 0..7
//   elements 8..15 -> k = (lane>>4)*16 + 8..15
// B 32x16 where B[k,n] = Brow[n,k] (Brow row-major, ld=K): lane&15 = col n;
//   elements 0..15 -> k = (lane>>4)*16 + 0..15   (one contiguous 32B run)
// Both use the identity K mapping -> dot product is exact.
// ---------------------------------------------------------------------------
DEV v16bf load_a_frag(const uint16_t* __restrict__ p) {
  v8bf lo = *(const v8bf*)(p);
  v8bf hi = *(const v8bf*)(p + 8);
  v16bf r;
#pragma unroll
  for (int i = 0; i < 8; ++i) { r[i] = lo[i]; r[i + 8] = hi[i]; }
  return r;
}
DEV v16bf load_b_frag(const uint16_t* __restrict__ p) {
  return *(const v16bf*)(p);
}

// ---------------------------------------------------------------------------
// Big GEMM:  out[i,j] = ((i==j) ? 1 : sum_k As[i,k]*Bs[j,k]) * adj[i,j]
// As, Bs: bf16 row-major [dim, K].  Block = 8 waves -> 128x64 tile,
// each wave 32x32 via 2x2 v_wmma_f32_16x16x32_bf16 accumulators.
// ---------------------------------------------------------------------------
__global__ __launch_bounds__(256)
void wmma_bf16_gram_mask(const uint16_t* __restrict__ As,
                         const uint16_t* __restrict__ Bs,
                         const float*    __restrict__ adj,
                         float*          __restrict__ out,
                         int dim, int K) {
  const int lane = threadIdx.x & 31;
  const int w    = threadIdx.x >> 5;          // 0..7
  const int row0 = blockIdx.y * 128 + (w >> 1) * 32;
  const int col0 = blockIdx.x * 64  + (w & 1) * 32;
  const int m    = lane & 15;
  const int hi   = lane >> 4;

  const uint16_t* pa0 = As + (size_t)(row0 + m) * K + hi * 16;
  const uint16_t* pa1 = pa0 + (size_t)16 * K;
  const uint16_t* pb0 = Bs + (size_t)(col0 + m) * K + hi * 16;
  const uint16_t* pb1 = pb0 + (size_t)16 * K;

  v8f c00 = {}, c01 = {}, c10 = {}, c11 = {};
  for (int k0 = 0; k0 < K; k0 += 32) {
    __builtin_prefetch(pa0 + k0 + 128, 0, 1);   // -> global_prefetch_b8
    __builtin_prefetch(pb0 + k0 + 128, 0, 1);
    v16bf a0 = load_a_frag(pa0 + k0);
    v16bf a1 = load_a_frag(pa1 + k0);
    v16bf b0 = load_b_frag(pb0 + k0);
    v16bf b1 = load_b_frag(pb1 + k0);
    c00 = __builtin_amdgcn_wmma_f32_16x16x32_bf16(false, a0, false, b0, (short)0, c00, false, false);
    c01 = __builtin_amdgcn_wmma_f32_16x16x32_bf16(false, a0, false, b1, (short)0, c01, false, false);
    c10 = __builtin_amdgcn_wmma_f32_16x16x32_bf16(false, a1, false, b0, (short)0, c10, false, false);
    c11 = __builtin_amdgcn_wmma_f32_16x16x32_bf16(false, a1, false, b1, (short)0, c11, false, false);
  }

  // Epilogue: C/D layout -> row = v + 8*(lane>>4), col = lane&15
  const int n  = lane & 15;
  const int mg = hi * 8;
  auto emit = [&](const v8f& c, int rb, int cb) {
#pragma unroll
    for (int v = 0; v < 8; ++v) {
      int i = rb + mg + v;
      int j = cb + n;
      size_t idx = (size_t)i * dim + j;
      float mult = (i == j) ? 1.0f : c[v];
      out[idx] = mult * adj[idx];
    }
  };
  emit(c00, row0,      col0);
  emit(c01, row0,      col0 + 16);
  emit(c10, row0 + 16, col0);
  emit(c11, row0 + 16, col0 + 16);
}

// ---------------------------------------------------------------------------
// Small GEMM + bias:  out[M,Nc] = A[M,K] @ B[K,Nc] + bias[Nc]   (all f32)
// Uses v_wmma_f32_16x16x4_f32 when available; plain VALU otherwise.
// Nc must be 16 or 64 for the WMMA path; grid.x = M / ((8/(Nc/16))*16).
// ---------------------------------------------------------------------------
__global__ __launch_bounds__(256)
void gemm_f32_bias(const float* __restrict__ A, const float* __restrict__ B,
                   const float* __restrict__ bias, float* __restrict__ out,
                   int M, int Nc, int K) {
#if HAVE_WMMA_F32
  const int lane   = threadIdx.x & 31;
  const int w      = threadIdx.x >> 5;
  const int tilesN = Nc >> 4;                  // 4 (Nc=64) or 1 (Nc=16)
  const int wc     = w % tilesN;
  const int wrs    = w / tilesN;
  const int rowsPB = (8 / tilesN) * 16;
  const int row0   = blockIdx.x * rowsPB + wrs * 16;
  const int col0   = wc * 16;
  const int m      = lane & 15;
  const int hk     = (lane >> 4) * 2;          // A 16x4: k = hk + {0,1}

  const float* pa = A + (size_t)(row0 + m) * K + hk;
  v8f acc = {};
#pragma unroll 4
  for (int k0 = 0; k0 < K; k0 += 4) {
    v2f a = *(const v2f*)(pa + k0);
    v2f b;
    b.x = B[(size_t)(k0 + hk)     * Nc + col0 + m];
    b.y = B[(size_t)(k0 + hk + 1) * Nc + col0 + m];
    acc = __builtin_amdgcn_wmma_f32_16x16x4_f32(false, a, false, b, (short)0, acc, false, false);
  }
  const int mg = (lane >> 4) * 8;
#pragma unroll
  for (int v = 0; v < 8; ++v)
    out[(size_t)(row0 + mg + v) * Nc + col0 + m] = acc[v] + bias[col0 + m];
#else
  size_t total = (size_t)M * Nc;
  for (size_t idx = (size_t)blockIdx.x * blockDim.x + threadIdx.x; idx < total;
       idx += (size_t)gridDim.x * blockDim.x) {
    int i = (int)(idx / Nc), c = (int)(idx % Nc);
    float s = 0.f;
    for (int k = 0; k < K; ++k) s += A[(size_t)i * K + k] * B[(size_t)k * Nc + c];
    out[idx] = s + bias[c];
  }
#endif
}

// ---------------------------------------------------------------------------
// Prep kernels
// ---------------------------------------------------------------------------
__global__ void compute_de(const float* __restrict__ H_e,
                           const float* __restrict__ p_v,
                           float* __restrict__ d_e) {
  int e = blockIdx.x * 256 + threadIdx.x;
  if (e < Ee) {
    float s = 0.f;
#pragma unroll
    for (int c = 0; c < DEIN; ++c) s += H_e[(size_t)e * DEIN + c] * p_v[c];
    d_e[e] = s;
  }
}

// Tb = bf16(T), Ts = bf16(T * d_e[col])    (column scaling = contraction dim)
__global__ void prep_T_bf16(const float* __restrict__ T,
                            const float* __restrict__ d_e,
                            uint16_t* __restrict__ Tb,
                            uint16_t* __restrict__ Ts) {
  int col = blockIdx.x * 256 + threadIdx.x;     // 0..E-1
  size_t idx = (size_t)blockIdx.y * Ee + col;   // row = blockIdx.y
  float t = T[idx];
  Tb[idx] = f2bf_raw(t);
  Ts[idx] = f2bf_raw(t * d_e[col]);
}

// Xv = H_v @ W_v  [N,64]; one block per row, 64 threads, row staged in LDS
__global__ void compute_Xv(const float* __restrict__ H_v,
                           const float* __restrict__ W_v,
                           float* __restrict__ Xv) {
  __shared__ float hrow[DVIN];
  int r = blockIdx.x, c = threadIdx.x;
  hrow[c]      = H_v[(size_t)r * DVIN + c];
  hrow[c + 64] = H_v[(size_t)r * DVIN + 64 + c];
  __syncthreads();
  float s = 0.f;
#pragma unroll 8
  for (int k = 0; k < DVIN; ++k) s += hrow[k] * W_v[(size_t)k * DVOUT + c];
  Xv[(size_t)r * DVOUT + c] = s;
}

__global__ void compute_dv(const float* __restrict__ Hv_out,
                           const float* __restrict__ p_e,
                           float* __restrict__ d_v) {
  int i = blockIdx.x * 256 + threadIdx.x;
  if (i < Nn) {
    float s = 0.f;
#pragma unroll 8
    for (int c = 0; c < DVOUT; ++c) s += Hv_out[(size_t)i * DVOUT + c] * p_e[c];
    d_v[i] = s;
  }
}

// Tt[e,n] = bf16(T[n,e]); Tts[e,n] = bf16(T[n,e]*d_v[n]) — LDS tiled transpose
__global__ void prep_Tt_bf16(const float* __restrict__ T,
                             const float* __restrict__ d_v,
                             uint16_t* __restrict__ Tt,
                             uint16_t* __restrict__ Tts) {
  __shared__ float tile[32][33];
  int eBase = blockIdx.x * 32;
  int nBase = blockIdx.y * 32;
  int tx = threadIdx.x;              // 0..31
  int ty = threadIdx.y;              // 0..7
  for (int r = ty; r < 32; r += 8)
    tile[r][tx] = T[(size_t)(nBase + r) * Ee + eBase + tx];
  __syncthreads();
  float dv = d_v[nBase + tx];
  for (int r = ty; r < 32; r += 8) {
    float v = tile[tx][r];           // = T[nBase+tx, eBase+r]
    size_t idx = (size_t)(eBase + r) * Nn + nBase + tx;
    Tt[idx]  = f2bf_raw(v);
    Tts[idx] = f2bf_raw(v * dv);
  }
}

// colmax[j] = max_i adjE[i,j]  (coalesced: thread j, loop rows)
__global__ void colmax_kernel(const float* __restrict__ adjE,
                              float* __restrict__ colmax) {
  int j = blockIdx.x * 256 + threadIdx.x;
  if (j < Ee) {
    float m = -__builtin_inff();
    for (int i = 0; i < Ee; ++i) m = fmaxf(m, adjE[(size_t)i * Ee + j]);
    colmax[j] = m;
  }
}

// Ys[j,c] = (H_e @ W_e)[j,c] / (colmax[j] + 1e-10)   (normalization folded in)
__global__ void compute_Ys(const float* __restrict__ H_e,
                           const float* __restrict__ W_e,
                           const float* __restrict__ colmax,
                           float* __restrict__ Ys) {
  int idx = blockIdx.x * 256 + threadIdx.x;
  if (idx < Ee * DEOUT) {
    int j = idx / DEOUT, c = idx % DEOUT;
    float s = 0.f;
#pragma unroll
    for (int k = 0; k < DEIN; ++k) s += H_e[(size_t)j * DEIN + k] * W_e[k * DEOUT + c];
    Ys[idx] = s / (colmax[j] + 1e-10f);
  }
}

// ---------------------------------------------------------------------------
// Launcher
// ---------------------------------------------------------------------------
extern "C" void kernel_launch(void* const* d_in, const int* in_sizes, int n_in,
                              void* d_out, int out_size, void* d_ws, size_t ws_size,
                              hipStream_t stream) {
  (void)in_sizes; (void)n_in; (void)out_size; (void)ws_size;
  const float* H_v   = (const float*)d_in[0];
  const float* H_e   = (const float*)d_in[1];
  const float* adj_v = (const float*)d_in[2];
  const float* adj_e = (const float*)d_in[3];
  const float* T     = (const float*)d_in[4];
  const float* W_v   = (const float*)d_in[5];
  const float* b_v   = (const float*)d_in[6];
  const float* p_v   = (const float*)d_in[7];
  const float* W_e   = (const float*)d_in[8];
  const float* b_e   = (const float*)d_in[9];
  const float* p_e   = (const float*)d_in[10];

  float* Hv_out = (float*)d_out;                             // [N, 64]
  float* He_out = (float*)d_out + (size_t)Nn * DVOUT;        // [E, 16]

  char*  ws  = (char*)d_ws;
  size_t off = 0;
  auto alloc = [&](size_t bytes) -> void* {
    void* p = ws + off;
    off = (off + bytes + 255) & ~(size_t)255;
    return p;
  };
  float*    d_e    = (float*)alloc((size_t)Ee * 4);
  float*    d_v    = (float*)alloc((size_t)Nn * 4);
  float*    Xv     = (float*)alloc((size_t)Nn * DVOUT * 4);
  float*    Ys     = (float*)alloc((size_t)Ee * DEOUT * 4);
  float*    colmax = (float*)alloc((size_t)Ee * 4);
  float*    adjVm  = (float*)alloc((size_t)Nn * Nn * 4);     // 16.8 MB
  float*    adjEm  = (float*)alloc((size_t)Ee * Ee * 4);     // 151 MB
  uint16_t* TbufA  = (uint16_t*)alloc((size_t)Nn * Ee * 2);  // Tb, reused as Tt
  uint16_t* TbufS  = (uint16_t*)alloc((size_t)Nn * Ee * 2);  // Ts, reused as Tts

  // ---- node layer ----
  compute_de<<<Ee / 256, 256, 0, stream>>>(H_e, p_v, d_e);
  prep_T_bf16<<<dim3(Ee / 256, Nn), 256, 0, stream>>>(T, d_e, TbufA, TbufS);
  compute_Xv<<<Nn, 64, 0, stream>>>(H_v, W_v, Xv);
  // adjVm = (diag->1 of Ts @ Tb^T) * adj_v     [N,N], K=E
  wmma_bf16_gram_mask<<<dim3(Nn / 64, Nn / 128), 256, 0, stream>>>(
      TbufS, TbufA, adj_v, adjVm, Nn, Ee);
  // Hv_out = adjVm @ Xv + b_v                  [N,64], K=N
  gemm_f32_bias<<<Nn / 32, 256, 0, stream>>>(adjVm, Xv, b_v, Hv_out, Nn, DVOUT, Nn);

  // ---- edge layer ----
  compute_dv<<<Nn / 256, 256, 0, stream>>>(Hv_out, p_e, d_v);
  prep_Tt_bf16<<<dim3(Ee / 32, Nn / 32), dim3(32, 8), 0, stream>>>(T, d_v, TbufA, TbufS);
  // adjEm = (diag->1 of Tts @ Tt^T) * adj_e    [E,E], K=N
  wmma_bf16_gram_mask<<<dim3(Ee / 64, Ee / 128), 256, 0, stream>>>(
      TbufS, TbufA, adj_e, adjEm, Ee, Nn);
  colmax_kernel<<<Ee / 256, 256, 0, stream>>>(adjEm, colmax);
  compute_Ys<<<(Ee * DEOUT) / 256, 256, 0, stream>>>(H_e, W_e, colmax, Ys);
  // He_out = adjEm @ Ys + b_e                  [E,16], K=E
  gemm_f32_bias<<<Ee / 128, 256, 0, stream>>>(adjEm, Ys, b_e, He_out, Ee, DEOUT, Ee);
}
